// LSTMLayerL2R_28656021799665
// MI455X (gfx1250) — compile-verified, matrix operand order checked by reading, not run on hardware
//
#include <hip/hip_runtime.h>
#include <hip/hip_bf16.h>

// LSTM: T=2048, B=32, D=512, H=512 (4H = 2048)
// Phase 1: pre[t,b,g] = x @ w_ih^T + b_ih  via v_wmma_f32_16x16x32_bf16
// Phase 2: persistent 16-WG recurrence, w_hh resident in LDS (bf16 B-fragments),
//          h kept in GLOBAL memory in WMMA A-fragment layout (double-buffered),
//          grid sync per timestep through an L2 atomic counter.

#define T_STEPS 2048
#define BATCH   32
#define DIM     512
#define HID     512
#define G4      2048                       // 4*H
#define NWG     16                         // recurrence workgroups
#define OUT_SEQ ((size_t)T_STEPS * BATCH * HID)
#define HFRAG_ELEMS (2 * 16 * 32 * 16)     // [mt][kt][lane][16] = 16384 bf16 = 32KB

typedef __attribute__((ext_vector_type(16))) __bf16 v16bf;
typedef __attribute__((ext_vector_type(8)))  float  v8f;

union Frag { v16bf v; __bf16 e[16]; uint4 q[2]; };

__device__ __forceinline__ float sigm_f(float x) {
    return 1.0f / (1.0f + __expf(-x));
}
__device__ __forceinline__ float tanh_f(float x) {
    x = fminf(fmaxf(x, -15.0f), 15.0f);
    float e = __expf(-2.0f * x);
    return (1.0f - e) / (1.0f + e);
}

// A-fragment (16x32 bf16) position of element (row m=b, k) in CDNA5 layout:
// lane L (half=L/16, mrow=L%16), element i -> k = kt*32 + (i/8)*16 + half*8 + i%8.
__device__ __forceinline__ int afrag_pos(int b, int k) {
    const int mt = b >> 4, mrow = b & 15;
    const int kt = k >> 5, ko = k & 31;
    const int group = ko >> 4, rem = ko & 15;
    const int half = rem >> 3, idx = rem & 7;
    const int lane = half * 16 + mrow;
    const int i = group * 8 + idx;
    return (((mt * 16 + kt) * 32) + lane) * 16 + i;
}

// ---------------------------------------------------------------------------
// Phase 1: pre-GEMM. M = T*B = 65536, N = 2048, K = 512.
// Block = 256 threads (8 waves). Block tile = 128(M) x 64(N).
// ---------------------------------------------------------------------------
__global__ __launch_bounds__(256) void lstm_pre_gemm(
    const float* __restrict__ x,      // [T*B, 512] row-major
    const float* __restrict__ w_ih,   // [2048, 512] row-major
    const float* __restrict__ b_ih,   // [2048]
    float* __restrict__ pre)          // [T*B, 2048]
{
    __shared__ __align__(32) __bf16 sB[4 * 32 * 16]; // [ntile][lane][16]

    const int tid  = threadIdx.x;
    const int wv   = tid >> 5;
    const int lane = tid & 31;
    const int half = lane >> 4;
    const int l16  = lane & 15;

    const size_t rowBase = (size_t)blockIdx.x * 128;
    const int    colBase = blockIdx.y * 64;

    v8f acc[4] = {};
    const size_t arow = rowBase + (size_t)wv * 16 + l16;

    for (int kt = 0; kt < 16; ++kt) {
        __syncthreads();
        if (tid < 128) {  // stage 4 B fragments
            const int nt  = tid >> 5;
            const int col = colBase + nt * 16 + l16;
            const float* wp = w_ih + (size_t)col * DIM + kt * 32 + half * 8;
            __bf16* dst = &sB[(nt * 32 + lane) * 16];
#pragma unroll
            for (int i = 0; i < 8; ++i) dst[i]     = (__bf16)wp[i];
#pragma unroll
            for (int i = 0; i < 8; ++i) dst[8 + i] = (__bf16)wp[16 + i];
        }
        __syncthreads();

        Frag fa;
        const float* xp = x + arow * DIM + kt * 32 + half * 8;
#pragma unroll
        for (int i = 0; i < 8; ++i) fa.e[i]     = (__bf16)xp[i];
#pragma unroll
        for (int i = 0; i < 8; ++i) fa.e[8 + i] = (__bf16)xp[16 + i];

#pragma unroll
        for (int nt = 0; nt < 4; ++nt) {
            v16bf fb = *(const v16bf*)&sB[(nt * 32 + lane) * 16];
            acc[nt] = __builtin_amdgcn_wmma_f32_16x16x32_bf16(
                false, fa.v, false, fb, (short)0, acc[nt], false, false);
        }
    }

#pragma unroll
    for (int nt = 0; nt < 4; ++nt) {
        const int col  = colBase + nt * 16 + l16;
        const float bias = b_ih[col];
#pragma unroll
        for (int r = 0; r < 8; ++r) {
            size_t row = rowBase + (size_t)wv * 16 + half * 8 + r;
            pre[row * (size_t)G4 + col] = acc[nt][r] + bias;
        }
    }
}

// ---------------------------------------------------------------------------
// Grid-wide split barrier via one L2 atomic counter (monotone targets).
// ---------------------------------------------------------------------------
__device__ __forceinline__ void grid_sync(int* cnt, int target, int tid) {
    __syncthreads();
    if (tid == 0) {
        __hip_atomic_fetch_add(cnt, 1, __ATOMIC_RELEASE, __HIP_MEMORY_SCOPE_AGENT);
        while (__hip_atomic_load(cnt, __ATOMIC_ACQUIRE, __HIP_MEMORY_SCOPE_AGENT) < target) {
            __builtin_amdgcn_s_sleep(2);
        }
    }
    __syncthreads();
    __threadfence();
}

__global__ void lstm_sync_init(int* s) { *s = 0; }

// ---------------------------------------------------------------------------
// Phase 2: persistent recurrence. 16 WGs x 256 threads. WG w owns hidden dims
// j in [32w, 32w+32)  ->  128 gate columns {i,f,g,o}x32 = 8 N-tiles.
// h lives in GLOBAL memory in A-fragment layout, double-buffered; A-fragments
// are loaded straight from L2 (global_load_b128), no per-step LDS staging.
// LDS carve (dynamic, 168448 B):
//   wfrag : 8nt*16kt*32*16 bf16 = 128 KB  (resident all steps)
//   pbuf  : 4*32b*32j fp32     =  16 KB  (pre[t] tile, prefetched per step)
//   gates : 4*32b*32j fp32     =  16 KB
//   cbuf  : 32b*32j fp32       =   4 KB
//   biasc : 128 fp32           =  512 B  (b_hh slice)
// ---------------------------------------------------------------------------
#define LDS_WFRAG 0
#define LDS_PBUF  (131072)
#define LDS_GATES (131072 + 16384)
#define LDS_CBUF  (131072 + 16384 + 16384)
#define LDS_BIAS  (131072 + 16384 + 16384 + 4096)
#define SMEM_BYTES (131072 + 16384 + 16384 + 4096 + 512)

__global__ __launch_bounds__(256) void lstm_recurrence(
    const float* __restrict__ pre,    // [T, B, 2048]
    const float* __restrict__ h0,     // [B, 512]
    const float* __restrict__ c0,     // [B, 512]
    const float* __restrict__ w_hh,   // [2048, 512]
    const float* __restrict__ b_hh,   // [2048]
    float* __restrict__ out,          // [T,B,H] then h[B,H] then c[B,H]
    __bf16* __restrict__ hbuf,        // [2][HFRAG_ELEMS] bf16, A-fragment layout
    int* __restrict__ sync)
{
    extern __shared__ __attribute__((aligned(32))) char smem[];
    __bf16* wfrag = (__bf16*)(smem + LDS_WFRAG);
    float*  pbuf  = (float*)(smem + LDS_PBUF);
    float*  gates = (float*)(smem + LDS_GATES);
    float*  cbuf  = (float*)(smem + LDS_CBUF);
    float*  biasc = (float*)(smem + LDS_BIAS);

    const int w    = blockIdx.x;
    const int tid  = threadIdx.x;
    const int wv   = tid >> 5;
    const int lane = tid & 31;
    const int half = lane >> 4;
    const int l16  = lane & 15;

    // Pack w_hh slice into B-fragment layout (once; resident for all steps).
    for (int ti = wv; ti < 128; ti += 8) {
        const int nt = ti >> 4, kt = ti & 15;
        const int c = nt * 16 + l16;
        const int gate = c >> 5, jloc = c & 31;
        const int g = gate * HID + w * 32 + jloc;
        const float* wp = w_hh + (size_t)g * HID + kt * 32 + half * 8;
        __bf16* dst = &wfrag[(size_t)(ti * 32 + lane) * 16];
#pragma unroll
        for (int i = 0; i < 8; ++i) dst[i]     = (__bf16)wp[i];
#pragma unroll
        for (int i = 0; i < 8; ++i) dst[8 + i] = (__bf16)wp[16 + i];
    }

    // Bias cache (b_hh folded into the pre-tile prefetch).
    if (tid < 128) {
        const int gate = tid >> 5, jl = tid & 31;
        biasc[tid] = b_hh[gate * HID + w * 32 + jl];
    }

    // Init c slice (LDS) and our slice of hbuf[0] (fragment positions).
    for (int e = tid; e < 1024; e += 256) {
        const int b = e >> 5, jl = e & 31;
        const int j = w * 32 + jl;
        cbuf[e] = c0[b * HID + j];
        hbuf[afrag_pos(b, j)] = (__bf16)h0[b * HID + j];
    }
    __syncthreads();
    __threadfence();

    int sidx = 0;
    grid_sync(sync, NWG * (++sidx), tid);   // all WGs published their h0 slice

    const int nt = wv;                       // this wave's N-tile
    const int cc = nt * 16 + l16;            // local gate column
    const int gate = cc >> 5, jloc = cc & 31;

    for (int t = 0; t < T_STEPS; ++t) {
        // --- Prefetch pre[t] tile (+bias) into LDS; overlaps with WMMA loop.
        {
            const int e0 = tid * 16;                 // 4096 floats / 256 threads
            const int pg = e0 >> 10;                 // gate
            const int pb = (e0 >> 5) & 31;           // batch
            const int pj = e0 & 31;                  // jloc (0 or 16)
            const float4* s4 = (const float4*)(pre + (size_t)t * BATCH * G4 +
                                               (size_t)pb * G4 + pg * HID + w * 32 + pj);
            float4* d4 = (float4*)&pbuf[e0];
#pragma unroll
            for (int q = 0; q < 4; ++q) {
                float4 v = s4[q];
                v.x += biasc[pg * 32 + pj + q * 4 + 0];
                v.y += biasc[pg * 32 + pj + q * 4 + 1];
                v.z += biasc[pg * 32 + pj + q * 4 + 2];
                v.w += biasc[pg * 32 + pj + q * 4 + 3];
                d4[q] = v;
            }
        }

        // --- Matmul: A-fragments straight from global (fragment-layout hbuf).
        const __bf16* hb = hbuf + (size_t)(t & 1) * HFRAG_ELEMS;
        v8f acc0 = {}, acc1 = {};
#pragma unroll
        for (int kt = 0; kt < 16; ++kt) {
            v16bf a0 = *(const v16bf*)(hb + (size_t)(((0 * 16 + kt) * 32) + lane) * 16);
            v16bf a1 = *(const v16bf*)(hb + (size_t)(((1 * 16 + kt) * 32) + lane) * 16);
            v16bf fb = *(const v16bf*)&wfrag[(size_t)(((nt * 16 + kt) * 32) + lane) * 16];
            acc0 = __builtin_amdgcn_wmma_f32_16x16x32_bf16(
                false, a0, false, fb, (short)0, acc0, false, false);
            acc1 = __builtin_amdgcn_wmma_f32_16x16x32_bf16(
                false, a1, false, fb, (short)0, acc1, false, false);
        }
        __syncthreads();   // pbuf ready + all waves done with their accs

        // --- Epilogue: gate pre-activations -> activations -> gates LDS.
#pragma unroll
        for (int r = 0; r < 8; ++r) {
            const int m = half * 8 + r;
            float v0 = acc0[r] + pbuf[gate * 1024 + m * 32 + jloc];         // batch m
            float v1 = acc1[r] + pbuf[gate * 1024 + (16 + m) * 32 + jloc];  // batch 16+m
            if (gate == 2) { v0 = tanh_f(v0); v1 = tanh_f(v1); }
            else           { v0 = sigm_f(v0); v1 = sigm_f(v1); }
            gates[gate * 1024 + m * 32 + jloc]        = v0;
            gates[gate * 1024 + (16 + m) * 32 + jloc] = v1;
        }
        __syncthreads();

        // --- Elementwise cell update; publish h_{t+1} into the OTHER buffer.
        __bf16* hn = hbuf + (size_t)((t + 1) & 1) * HFRAG_ELEMS;
        for (int e = tid; e < 1024; e += 256) {
            const int b = e >> 5, jl = e & 31;
            const float iv = gates[e];
            const float fv = gates[1024 + e];
            const float gv = gates[2048 + e];
            const float ov = gates[3072 + e];
            const float cn = fv * cbuf[e] + iv * gv;
            cbuf[e] = cn;
            const float hv = ov * tanh_f(cn);
            const int j = w * 32 + jl;
            out[((size_t)t * BATCH + b) * HID + j] = hv;
            hn[afrag_pos(b, j)] = (__bf16)hv;
        }
        grid_sync(sync, NWG * (++sidx), tid);
    }

    // Final h (copy of last output row) and final c.
    for (int e = tid; e < 1024; e += 256) {
        const int b = e >> 5, jl = e & 31;
        const int j = w * 32 + jl;
        out[OUT_SEQ + (size_t)b * HID + j] =
            out[((size_t)(T_STEPS - 1) * BATCH + b) * HID + j];
        out[OUT_SEQ + (size_t)BATCH * HID + (size_t)b * HID + j] = cbuf[e];
    }
}

// ---------------------------------------------------------------------------
// Launch. Workspace: pre (512 MiB fp32) | hbuf 2x32KB (bf16 frags) | sync (int)
// ---------------------------------------------------------------------------
extern "C" void kernel_launch(void* const* d_in, const int* in_sizes, int n_in,
                              void* d_out, int out_size, void* d_ws, size_t ws_size,
                              hipStream_t stream) {
    const float* x    = (const float*)d_in[0];
    const float* h0   = (const float*)d_in[1];
    const float* c0   = (const float*)d_in[2];
    const float* w_ih = (const float*)d_in[3];
    const float* w_hh = (const float*)d_in[4];
    const float* b_ih = (const float*)d_in[5];
    const float* b_hh = (const float*)d_in[6];
    float* out = (float*)d_out;

    char* ws = (char*)d_ws;
    const size_t preBytes = (size_t)T_STEPS * BATCH * G4 * sizeof(float); // 512 MiB
    float*  pre  = (float*)ws;
    __bf16* hbuf = (__bf16*)(ws + preBytes);                // 2 * 32 KB
    int*    sync = (int*)(ws + preBytes + 2 * HFRAG_ELEMS * sizeof(__bf16));

    (void)hipFuncSetAttribute(reinterpret_cast<const void*>(lstm_recurrence),
                              hipFuncAttributeMaxDynamicSharedMemorySize, SMEM_BYTES);

    lstm_sync_init<<<1, 1, 0, stream>>>(sync);
    lstm_pre_gemm<<<dim3(512, 32), 256, 0, stream>>>(x, w_ih, b_ih, pre);
    lstm_recurrence<<<NWG, 256, SMEM_BYTES, stream>>>(
        pre, h0, c0, w_hh, b_hh, out, hbuf, sync);
}
